// GlobalAttn_11003706212375
// MI455X (gfx1250) — compile-verified, matrix operand order checked by reading, not run on hardware
//
#include <hip/hip_runtime.h>
#include <math.h>

// ---------------- problem constants (match reference) ----------------
#define EMB       256
#define TWO_EMB   512
#define NHEAD     4
#define HDIM      64
#define NEDGE     320000
#define NNODE     10000
#define SOFT_EPS  1e-16f

typedef __attribute__((ext_vector_type(16))) __bf16 v16bf;
typedef __attribute__((ext_vector_type(8)))  __bf16 v8bf;
typedef __attribute__((ext_vector_type(8)))  float  v8f;

// monotone float->uint map so unsigned atomicMax == float max
__device__ __forceinline__ unsigned f2ord(float f) {
  unsigned u = __float_as_uint(f);
  return (u & 0x80000000u) ? ~u : (u | 0x80000000u);
}
__device__ __forceinline__ float ord2f(unsigned u) {
  unsigned v = (u & 0x80000000u) ? (u & 0x7fffffffu) : ~u;
  return __uint_as_float(v);
}
#define ORD_NEG_INF 0x007fffffu  // f2ord(-inf)

// ---------------- pass 0a: convert w1 (fp32 [256][512]) -> bf16 ----------------
__global__ void cvt_w1_kernel(const float* __restrict__ w, __bf16* __restrict__ o) {
  int i = blockIdx.x * 256 + threadIdx.x;
  if (i < EMB * TWO_EMB) o[i] = (__bf16)w[i];
}

// ---------------- pass 0b: init segment max/sum scratch ----------------
__global__ void init_seg_kernel(unsigned* __restrict__ segmax, float* __restrict__ segsum) {
  int i = blockIdx.x * 256 + threadIdx.x;
  if (i < NNODE * NHEAD) { segmax[i] = ORD_NEG_INF; segsum[i] = 0.0f; }
}

// B fragment: 16 bf16 per lane (lo = K c0..c0+7, hi = K c0+16..c0+23 pattern)
struct BF { v8bf lo, hi; };

__device__ __forceinline__ BF loadB(const __bf16* __restrict__ brow, int t, int half) {
  const int ks = t >> 4, n = t & 15;
  const __bf16* p = brow + (size_t)n * 16 * TWO_EMB + ks * 32 + half * 8;
  BF f;
  f.lo = *(const v8bf*)(p);
  f.hi = *(const v8bf*)(p + 16);
  return f;
}

// ---------------- pass 1: fused  x=[q|k]@w1^T + b -> leaky -> per-head dot ----
// one wave32 per 16-edge tile; WMMA f32 = bf16 A[16x32] * bf16 B[32x16]
// Software-pipelined: B depth-4 register pipeline, A raw prefetched 1 k-step ahead.
__global__ __launch_bounds__(128) void attn_score_kernel(
    const float* __restrict__ q, const float* __restrict__ k,
    const __bf16* __restrict__ bw1,      // [256][512] bf16 (row n = output feature)
    const float*  __restrict__ w1b,      // [256]
    const float*  __restrict__ w2,       // [4][64] -> flat index == feature index
    float* __restrict__ attn)            // [E][4]
{
  const int wave = threadIdx.x >> 5;
  const int lane = threadIdx.x & 31;
  const int tile = blockIdx.x * 4 + wave;        // 20000 tiles
  const int m    = lane & 15;                    // row within tile / N within ntile
  const int half = lane >> 4;                    // K-chunk select per ISA layout
  const int e0   = tile * 16;

  const float* __restrict__ qrow = q + (size_t)(e0 + m) * EMB;
  const float* __restrict__ krow = k + (size_t)(e0 + m) * EMB;

  v8f acc[16];
  #pragma unroll
  for (int n = 0; n < 16; ++n) acc[n] = (v8f){0.f,0.f,0.f,0.f,0.f,0.f,0.f,0.f};

  const __bf16* __restrict__ brow = bw1 + (size_t)m * TWO_EMB;

  // ---- prologue: A raw for ks=0, B pipeline depth 4 ----
  const int cA0 = half * 8;        // ks=0: c0
  const int cA1 = cA0 + 16;        // ks=0: c1
  float4 ar0 = *(const float4*)(qrow + cA0);
  float4 ar1 = *(const float4*)(qrow + cA0 + 4);
  float4 ar2 = *(const float4*)(qrow + cA1);
  float4 ar3 = *(const float4*)(qrow + cA1 + 4);

  BF b0 = loadB(brow, 0, half);
  BF b1 = loadB(brow, 1, half);
  BF b2 = loadB(brow, 2, half);
  BF b3 = loadB(brow, 3, half);

  for (int ks = 0; ks < 16; ++ks) {
    // convert current A raw -> bf16 fragment
    v16bf A;
    A[0]=(__bf16)ar0.x;  A[1]=(__bf16)ar0.y;  A[2]=(__bf16)ar0.z;  A[3]=(__bf16)ar0.w;
    A[4]=(__bf16)ar1.x;  A[5]=(__bf16)ar1.y;  A[6]=(__bf16)ar1.z;  A[7]=(__bf16)ar1.w;
    A[8]=(__bf16)ar2.x;  A[9]=(__bf16)ar2.y;  A[10]=(__bf16)ar2.z; A[11]=(__bf16)ar2.w;
    A[12]=(__bf16)ar3.x; A[13]=(__bf16)ar3.y; A[14]=(__bf16)ar3.z; A[15]=(__bf16)ar3.w;

    // prefetch A raw for next k-step (clamped; redundant reload on last step)
    {
      const int ksn = (ks < 15) ? (ks + 1) : 15;
      const int c0 = ksn * 32 + half * 8;
      const int c1 = c0 + 16;
      const float* r0 = (c0 < EMB) ? (qrow + c0) : (krow + (c0 - EMB));
      const float* r1 = (c1 < EMB) ? (qrow + c1) : (krow + (c1 - EMB));
      ar0 = *(const float4*)(r0);
      ar1 = *(const float4*)(r0 + 4);
      ar2 = *(const float4*)(r1);
      ar3 = *(const float4*)(r1 + 4);
    }

    #pragma unroll
    for (int n = 0; n < 16; ++n) {
      BF cur = b0; b0 = b1; b1 = b2; b2 = b3;           // renamed away by unroll
      int t = ks * 16 + n + 4;
      if (t > 255) t = 255;                              // clamp: harmless reload
      b3 = loadB(brow, t, half);

      v16bf B = __builtin_shufflevector(cur.lo, cur.hi,
                  0,1,2,3,4,5,6,7,8,9,10,11,12,13,14,15);
      acc[n] = __builtin_amdgcn_wmma_f32_16x16x32_bf16(
          /*neg_a=*/false, A, /*neg_b=*/false, B,
          /*c_mod=*/(short)0, acc[n], /*reuse_a=*/false, /*reuse_b=*/false);
    }
  }

  // epilogue: bias + leaky_relu(0.01) + per-head dot with w2, per-lane partials
  float hsum[NHEAD][8];
  #pragma unroll
  for (int h = 0; h < NHEAD; ++h)
    #pragma unroll
    for (int r = 0; r < 8; ++r) hsum[h][r] = 0.0f;

  #pragma unroll
  for (int n = 0; n < 16; ++n) {
    const int f   = n * 16 + m;          // output feature held by this lane
    const float b = w1b[f];
    const float wv = w2[f];              // [4][64] flat == f
    const int h = n >> 2;
    #pragma unroll
    for (int r = 0; r < 8; ++r) {
      float x = acc[n][r] + b;
      x = (x > 0.0f) ? x : 0.01f * x;
      hsum[h][r] += x * wv;
    }
  }

  // reduce the 16 N-lanes (lanes 0-15 and 16-31 are independent M groups)
  #pragma unroll
  for (int mask = 1; mask <= 8; mask <<= 1)
    #pragma unroll
    for (int h = 0; h < NHEAD; ++h)
      #pragma unroll
      for (int r = 0; r < 8; ++r)
        hsum[h][r] += __shfl_xor(hsum[h][r], mask, 32);

  if (m == 0) {  // lane 0 -> edges e0..e0+7 (reg r = M=r); lane 16 -> e0+8..e0+15
    const int ebase = e0 + half * 8;
    #pragma unroll
    for (int r = 0; r < 8; ++r) {
      float4 v = make_float4(hsum[0][r], hsum[1][r], hsum[2][r], hsum[3][r]);
      *(float4*)(attn + (size_t)(ebase + r) * NHEAD) = v;
    }
  }
}

// ---------------- pass 2: segment max (atomic on ordered uints) ----------------
__global__ void seg_max_kernel(const float* __restrict__ attn, const int* __restrict__ idx,
                               unsigned* __restrict__ segmax) {
  int i = blockIdx.x * 256 + threadIdx.x;
  if (i >= NEDGE * NHEAD) return;
  int e = i >> 2, h = i & 3;
  atomicMax(&segmax[idx[e] * NHEAD + h], f2ord(attn[i]));
}

// ---------------- pass 3: exp(a - max) -> out, atomic segment sum ----------------
__global__ void seg_exp_kernel(const float* __restrict__ attn, const int* __restrict__ idx,
                               const unsigned* __restrict__ segmax,
                               float* __restrict__ segsum, float* __restrict__ out) {
  int i = blockIdx.x * 256 + threadIdx.x;
  if (i >= NEDGE * NHEAD) return;
  int e = i >> 2, h = i & 3;
  int s = idx[e] * NHEAD + h;
  float v = expf(attn[i] - ord2f(segmax[s]));
  out[i] = v;
  atomicAdd(&segsum[s], v);
}

// ---------------- pass 4: normalize ----------------
__global__ void seg_norm_kernel(const int* __restrict__ idx,
                                const float* __restrict__ segsum, float* __restrict__ out) {
  int i = blockIdx.x * 256 + threadIdx.x;
  if (i >= NEDGE * NHEAD) return;
  int e = i >> 2, h = i & 3;
  out[i] /= (segsum[idx[e] * NHEAD + h] + SOFT_EPS);
}

// ---------------- host launcher ----------------
extern "C" void kernel_launch(void* const* d_in, const int* in_sizes, int n_in,
                              void* d_out, int out_size, void* d_ws, size_t ws_size,
                              hipStream_t stream) {
  const float* q    = (const float*)d_in[0];
  const float* kk   = (const float*)d_in[1];
  const float* w1w  = (const float*)d_in[2];
  const float* w1b  = (const float*)d_in[3];
  const float* w2   = (const float*)d_in[4];
  const int*   idx  = (const int*)d_in[5];
  float* out = (float*)d_out;

  // scratch layout
  char* ws = (char*)d_ws;
  __bf16*   bw1    = (__bf16*)ws;                                  // 262144 B
  float*    attn   = (float*)(ws + 262144);                        // E*4*4 = 5120000 B
  unsigned* segmax = (unsigned*)(ws + 262144 + 5120000);           // 160000 B
  float*    segsum = (float*)(ws + 262144 + 5120000 + 160000);     // 160000 B

  cvt_w1_kernel<<<(EMB * TWO_EMB + 255) / 256, 256, 0, stream>>>(w1w, bw1);
  init_seg_kernel<<<(NNODE * NHEAD + 255) / 256, 256, 0, stream>>>(segmax, segsum);

  // 20000 edge tiles, 4 waves/block
  attn_score_kernel<<<5000, 128, 0, stream>>>(q, kk, bw1, w1b, w2, attn);

  const int EH = NEDGE * NHEAD;
  seg_max_kernel<<<(EH + 255) / 256, 256, 0, stream>>>(attn, idx, segmax);
  seg_exp_kernel<<<(EH + 255) / 256, 256, 0, stream>>>(attn, idx, segmax, segsum, out);
  seg_norm_kernel<<<(EH + 255) / 256, 256, 0, stream>>>(idx, segsum, out);
}
